// GraphSAGE_24189255811506
// MI455X (gfx1250) — matrix operand order repaired and checked by reading, not compiled, not run on hardware
//
#include <hip/hip_runtime.h>

// ---------------------------------------------------------------------------
// GraphSAGE on MI455X (gfx1250, wave32). Layer-2 GEMMs via v_wmma_f32_16x16x32_bf16.
// ---------------------------------------------------------------------------

#define NN 100000
#define NE 1600000
#define NG 5000
#define HID 256

typedef __bf16 bf16_t;
typedef __attribute__((ext_vector_type(16))) __bf16 bf16x16;
typedef __attribute__((ext_vector_type(8)))  __bf16 bf16x8;
typedef __attribute__((ext_vector_type(8)))  float  f32x8;

// ---------------------------------------------------------------- zero
__global__ void zero_kernel(float4* __restrict__ p, size_t n4) {
    size_t i = (size_t)blockIdx.x * blockDim.x + threadIdx.x;
    if (i < n4) p[i] = make_float4(0.f, 0.f, 0.f, 0.f);
}

// ---------------------------------------------------------------- pack W2{l,r} into WMMA B layout
// Packed layout: for (jt, kt, lane): 16 contiguous bf16 = W[kt*32 + (i>>3)*16 + (lane>>4)*8 + (i&7)][jt*16 + (lane&15)]
__global__ void pack_w2_kernel(const float* __restrict__ W2l,
                               const float* __restrict__ W2r,
                               bf16_t* __restrict__ Pl,
                               bf16_t* __restrict__ Pr) {
    int tid = blockIdx.x * blockDim.x + threadIdx.x;   // 0 .. 2*65536-1
    int w    = tid >> 16;
    int idx  = tid & 65535;
    int i    = idx & 15;
    int lane = (idx >> 4) & 31;
    int kt   = (idx >> 9) & 7;
    int jt   = (idx >> 12) & 15;
    int k = kt * 32 + (i >> 3) * 16 + (lane >> 4) * 8 + (i & 7);
    int j = jt * 16 + (lane & 15);
    const float* W = w ? W2r : W2l;
    bf16_t*      P = w ? Pr  : Pl;
    P[idx] = (bf16_t)W[k * HID + j];
}

// ---------------------------------------------------------------- degree / graph counts
__global__ void degree_kernel(const int* __restrict__ dst, float* __restrict__ cnt) {
    size_t e = (size_t)blockIdx.x * blockDim.x + threadIdx.x;
    if (e < NE) unsafeAtomicAdd(&cnt[dst[e]], 1.0f);
}

__global__ void gcnt_kernel(const int* __restrict__ batch, float* __restrict__ gcnt) {
    size_t n = (size_t)blockIdx.x * blockDim.x + threadIdx.x;
    if (n < NN) unsafeAtomicAdd(&gcnt[batch[n]], 1.0f);
}

// ---------------------------------------------------------------- layer-1 edge aggregation (14 feats)
__global__ void agg1_kernel(const int* __restrict__ src, const int* __restrict__ dst,
                            const float* __restrict__ x, const float* __restrict__ pos,
                            float* __restrict__ agg1) {
    size_t tid = (size_t)blockIdx.x * blockDim.x + threadIdx.x;
    size_t e = tid >> 4;
    int    k = (int)(tid & 15);
    if (e < NE && k < 14) {
        int s = src[e], d = dst[e];
        float v = (k < 11) ? x[(size_t)s * 11 + k] : pos[(size_t)s * 3 + (k - 11)];
        unsafeAtomicAdd(&agg1[(size_t)d * 14 + k], v);
    }
}

// ---------------------------------------------------------------- layer-1 dense (K=14): h1 = relu(mean@W1l + b1 + h0@W1r)
__global__ void layer1_kernel(const float* __restrict__ agg1, const float* __restrict__ cnt,
                              const float* __restrict__ x, const float* __restrict__ pos,
                              const float* __restrict__ W1l, const float* __restrict__ b1,
                              const float* __restrict__ W1r,
                              bf16_t* __restrict__ h1bf) {
    __shared__ float s_feat[28];   // [0..13] = mean-aggregated, [14..27] = root h0
    int n = blockIdx.x;
    int t = threadIdx.x;
    if (t < 14) {
        float c = cnt[n]; c = c > 1.0f ? c : 1.0f;
        s_feat[t] = agg1[(size_t)n * 14 + t] / c;
    } else if (t < 28) {
        int k = t - 14;
        s_feat[t] = (k < 11) ? x[(size_t)n * 11 + k] : pos[(size_t)n * 3 + (k - 11)];
    }
    __syncthreads();
    float acc = b1[t];
#pragma unroll
    for (int k = 0; k < 14; ++k) {
        acc += s_feat[k] * W1l[k * HID + t];
        acc += s_feat[14 + k] * W1r[k * HID + t];
    }
    acc = acc > 0.f ? acc : 0.f;
    h1bf[(size_t)n * HID + t] = (bf16_t)acc;
}

// ---------------------------------------------------------------- layer-2 edge aggregation (256 feats, from bf16)
__global__ void agg2_kernel(const int* __restrict__ src, const int* __restrict__ dst,
                            const bf16_t* __restrict__ h1bf, float* __restrict__ agg2) {
    size_t tid = (size_t)blockIdx.x * blockDim.x + threadIdx.x;   // NE * 64 threads
    size_t e = tid >> 6;
    int    c = (int)(tid & 63) << 2;
    if (e < NE) {
        int s = src[e], d = dst[e];
        const bf16_t* hp = h1bf + (size_t)s * HID + c;
        float*        ap = agg2 + (size_t)d * HID + c;
#pragma unroll
        for (int i = 0; i < 4; ++i) unsafeAtomicAdd(ap + i, (float)hp[i]);
    }
}

// ---------------------------------------------------------------- mean + convert to bf16
__global__ void mean2_kernel(const float* __restrict__ agg2, const float* __restrict__ cnt,
                             bf16_t* __restrict__ mean2bf) {
    size_t tid = (size_t)blockIdx.x * blockDim.x + threadIdx.x;   // NN * 64 threads
    size_t n = tid >> 6;
    int    c = (int)(tid & 63) << 2;
    if (n < NN) {
        float cc = cnt[n]; cc = cc > 1.0f ? cc : 1.0f;
        float inv = 1.0f / cc;
#pragma unroll
        for (int i = 0; i < 4; ++i)
            mean2bf[n * HID + c + i] = (bf16_t)(agg2[n * HID + c + i] * inv);
    }
}

// ---------------------------------------------------------------- WMMA A-tile load (16-bit A 16x32 layout)
// lanes 0-15: row M=lane,   K = {0..7} and {16..23} of the k-step
// lanes 16-31: row M=lane-16, K = {8..15} and {24..31}
__device__ __forceinline__ bf16x16 load_a_tile(const bf16_t* __restrict__ rowptr, int kbase, int hi) {
    const bf16x8 lo = *reinterpret_cast<const bf16x8*>(rowptr + kbase + hi * 8);
    const bf16x8 hv = *reinterpret_cast<const bf16x8*>(rowptr + kbase + 16 + hi * 8);
    bf16x16 r;
#pragma unroll
    for (int i = 0; i < 8; ++i) { r[i] = lo[i]; r[i + 8] = hv[i]; }
    return r;
}

// ---------------------------------------------------------------- fused layer-2 GEMM:
// h2 = relu(mean2 @ W2l + b2 + h1 @ W2r), all 16x16x32 bf16 WMMA, f32 accum.
// Block = 256 threads = 8 waves; wave (w) -> mtile = 2*blockIdx.x + (w>>2), quarter = w&3 (64 cols).
__global__ void gemm2_wmma_kernel(const bf16_t* __restrict__ mean2bf,
                                  const bf16_t* __restrict__ h1bf,
                                  const bf16_t* __restrict__ wlp,
                                  const bf16_t* __restrict__ wrp,
                                  const float* __restrict__ b2,
                                  float* __restrict__ h2) {
    const int lane    = threadIdx.x & 31;
    const int wave    = threadIdx.x >> 5;
    const int mtile   = blockIdx.x * 2 + (wave >> 2);
    const int quarter = wave & 3;
    const int m0      = mtile * 16;
    const int hi      = lane >> 4;
    const int l15     = lane & 15;

    const bf16_t* arowM = mean2bf + (size_t)(m0 + l15) * HID;
    const bf16_t* arowH = h1bf    + (size_t)(m0 + l15) * HID;

    f32x8 acc[4];
#pragma unroll
    for (int nt = 0; nt < 4; ++nt) {
        float bj = b2[(quarter * 4 + nt) * 16 + l15];
#pragma unroll
        for (int r = 0; r < 8; ++r) acc[nt][r] = bj;
    }

#pragma unroll
    for (int kt = 0; kt < 8; ++kt) {
        if (kt < 7) {
            __builtin_prefetch(arowM + (kt + 1) * 32, 0, 0);
            __builtin_prefetch(arowH + (kt + 1) * 32, 0, 0);
        }
        const bf16x16 aM = load_a_tile(arowM, kt * 32, hi);
        const bf16x16 aH = load_a_tile(arowH, kt * 32, hi);
#pragma unroll
        for (int nt = 0; nt < 4; ++nt) {
            const int jt = quarter * 4 + nt;
            const size_t boff = ((size_t)(jt * 8 + kt) * 32 + lane) * 16;
            const bf16x16 bL = *reinterpret_cast<const bf16x16*>(wlp + boff);
            acc[nt] = __builtin_amdgcn_wmma_f32_16x16x32_bf16(
                false, aM, false, bL, (short)0, acc[nt], false, false);
            const bf16x16 bR = *reinterpret_cast<const bf16x16*>(wrp + boff);
            acc[nt] = __builtin_amdgcn_wmma_f32_16x16x32_bf16(
                false, aH, false, bR, (short)0, acc[nt], false, false);
        }
    }

    // C/D f32 layout: VGPR r, lanes 0-15 -> M=r, lanes 16-31 -> M=r+8; N = lane&15
#pragma unroll
    for (int nt = 0; nt < 4; ++nt) {
        const int j = (quarter * 4 + nt) * 16 + l15;
#pragma unroll
        for (int r = 0; r < 8; ++r) {
            float v = acc[nt][r];
            v = v > 0.f ? v : 0.f;
            h2[(size_t)(m0 + r + hi * 8) * HID + j] = v;
        }
    }
}

// ---------------------------------------------------------------- global mean pool (sum part)
__global__ void pool_kernel(const float* __restrict__ h2, const int* __restrict__ batch,
                            float* __restrict__ gsum) {
    size_t tid = (size_t)blockIdx.x * blockDim.x + threadIdx.x;   // NN * 64
    size_t n = tid >> 6;
    int    c = (int)(tid & 63) << 2;
    if (n < NN) {
        int g = batch[n];
        const float* hp = h2 + n * HID + c;
        float* gp = gsum + (size_t)g * HID + c;
#pragma unroll
        for (int i = 0; i < 4; ++i) unsafeAtomicAdd(gp + i, hp[i]);
    }
}

// ---------------------------------------------------------------- final: out[g] = (gsum[g]/cnt) @ Wout + bout
__global__ void out_kernel(const float* __restrict__ gsum, const float* __restrict__ gcnt,
                           const float* __restrict__ Wout, const float* __restrict__ bout,
                           float* __restrict__ out) {
    __shared__ float red[8];
    const int g = blockIdx.x;
    const int t = threadIdx.x;
    float cg = gcnt[g]; cg = cg > 1.0f ? cg : 1.0f;
    float v = (gsum[(size_t)g * HID + t] / cg) * Wout[t];
#pragma unroll
    for (int o = 16; o > 0; o >>= 1) v += __shfl_down(v, o, 32);
    if ((t & 31) == 0) red[t >> 5] = v;
    __syncthreads();
    if (t == 0) {
        float s = 0.f;
#pragma unroll
        for (int i = 0; i < 8; ++i) s += red[i];
        out[g] = s + bout[0];
    }
}

// ---------------------------------------------------------------------------
extern "C" void kernel_launch(void* const* d_in, const int* in_sizes, int n_in,
                              void* d_out, int out_size, void* d_ws, size_t ws_size,
                              hipStream_t stream) {
    (void)in_sizes; (void)n_in; (void)out_size; (void)ws_size;

    const float* x    = (const float*)d_in[0];
    const float* pos  = (const float*)d_in[1];
    const int*   ei   = (const int*)d_in[2];
    const int*   batch= (const int*)d_in[3];
    const float* W1l  = (const float*)d_in[4];
    const float* b1   = (const float*)d_in[5];
    const float* W1r  = (const float*)d_in[6];
    const float* W2l  = (const float*)d_in[7];
    const float* b2   = (const float*)d_in[8];
    const float* W2r  = (const float*)d_in[9];
    const float* Wout = (const float*)d_in[10];
    const float* bout = (const float*)d_in[11];
    float* out = (float*)d_out;

    const int* src = ei;
    const int* dst = ei + NE;

    // ---- workspace carve-up (256B aligned); zeroed region first
    char* ws = (char*)d_ws;
    size_t off = 0;
    auto carve = [&](size_t bytes) -> void* {
        void* p = ws + off;
        off = (off + bytes + 255) & ~(size_t)255;
        return p;
    };
    float*  cnt     = (float*)carve((size_t)NN * 4);
    float*  gcnt    = (float*)carve((size_t)NG * 4);
    float*  agg1    = (float*)carve((size_t)NN * 14 * 4);
    float*  agg2    = (float*)carve((size_t)NN * HID * 4);
    float*  gsum    = (float*)carve((size_t)NG * HID * 4);
    const size_t zero_bytes = off;                       // everything above starts at 0
    bf16_t* h1bf    = (bf16_t*)carve((size_t)NN * HID * 2);
    bf16_t* mean2bf = (bf16_t*)carve((size_t)NN * HID * 2);
    bf16_t* w2lp    = (bf16_t*)carve((size_t)65536 * 2);
    bf16_t* w2rp    = (bf16_t*)carve((size_t)65536 * 2);
    float*  h2      = agg2;   // agg2 is dead after mean2_kernel -> reuse for h2

    const int TPB = 256;
    auto blocks = [](size_t n, int tpb) { return (unsigned)((n + tpb - 1) / tpb); };

    // 1. zero accumulators
    size_t n4 = zero_bytes / 16;
    zero_kernel<<<blocks(n4, TPB), TPB, 0, stream>>>((float4*)ws, n4);

    // 2. pack layer-2 weights into WMMA B layout (bf16)
    pack_w2_kernel<<<blocks(2 * 65536, TPB), TPB, 0, stream>>>(W2l, W2r, w2lp, w2rp);

    // 3. degree + graph counts
    degree_kernel<<<blocks(NE, TPB), TPB, 0, stream>>>(dst, cnt);
    gcnt_kernel<<<blocks(NN, TPB), TPB, 0, stream>>>(batch, gcnt);

    // 4. layer-1: edge aggregation then dense + relu -> h1 (bf16)
    agg1_kernel<<<blocks((size_t)NE * 16, TPB), TPB, 0, stream>>>(src, dst, x, pos, agg1);
    layer1_kernel<<<NN, HID, 0, stream>>>(agg1, cnt, x, pos, W1l, b1, W1r, h1bf);

    // 5. layer-2: edge aggregation, mean, fused WMMA GEMMs + bias + relu -> h2
    agg2_kernel<<<blocks((size_t)NE * 64, TPB), TPB, 0, stream>>>(src, dst, h1bf, agg2);
    mean2_kernel<<<blocks((size_t)NN * 64, TPB), TPB, 0, stream>>>(agg2, cnt, mean2bf);
    // 6250 m-tiles of 16 rows, 2 m-tiles per block (8 waves: 2 m x 4 col-quarters)
    gemm2_wmma_kernel<<<6250 / 2, TPB, 0, stream>>>(mean2bf, h1bf, w2lp, w2rp, b2, h2);

    // 6. global mean pool + output head
    pool_kernel<<<blocks((size_t)NN * 64, TPB), TPB, 0, stream>>>(h2, batch, gsum);
    out_kernel<<<NG, HID, 0, stream>>>(gsum, gcnt, Wout, bout, out);
}
